// NCutLoss_52767968198856
// MI455X (gfx1250) — matrix operand order repaired and checked by reading, not compiled, not run on hardware
//
#include <hip/hip_runtime.h>
#include <hip/hip_bf16.h>

// NCut loss for MI455X (gfx1250, wave32).
// v2: fragment staging in WMMA-native per-lane layout (2x ds_load_b128 per
// operand instead of 16x ds_load_u16 + v_mov_b16 merges), 48B-padded class
// halo rows so the hot 69-offset loop uses 16B vector LDS loads, interior /
// border loop specialization, and direct global-atomic Gram flush (no LDS
// num accumulator) to fit the padded halos in 64KB LDS.

typedef __attribute__((ext_vector_type(16))) _Float16 v16h;
typedef __attribute__((ext_vector_type(8)))  _Float16 h8;
typedef __attribute__((ext_vector_type(8)))  float    v8f;

#define NC   21      // classes
#define CF   32      // feature channels
#define CS   24      // padded class-row stride in halves (48 B, 16B aligned)
#define HH   256
#define WW   256
#define BB   8
#define TILE 16
#define HALO 4       // radius 5 -> offsets |dy|,|dx| <= 4
#define HT   (TILE + 2*HALO)   // 24
#define KPAD 32      // classes padded for WMMA (col 21 = degree)

// LDS layout (64,512 B total):
//   phase 1: [0, 27648)        class halo, f16, [576][24]
//            [27648, 64512)    feature halo, f16, [576][32]
//   phase 2 (overlay): [0,16384)      sAfrag  [8 waves][2 moT][32 lanes][16 halves]
//                      [16384,32768)  sBfrag  [8 waves][2 noT][32 lanes][16 halves]
#define CLS_BYTES  (HT*HT*CS*2)          // 27648
#define FEAT_OFF   CLS_BYTES
#define FEAT_BYTES (HT*HT*CF*2)          // 36864
#define SMEM_BYTES (CLS_BYTES + FEAT_BYTES)   // 64512

__global__ void ncut_zero_kernel(float* __restrict__ gnum, int n) {
  int i = blockIdx.x * blockDim.x + threadIdx.x;
  if (i < n) gnum[i] = 0.0f;
}

template <bool CHECK>
__device__ __forceinline__ void affinity_loop(
    const _Float16* __restrict__ featLDS, const _Float16* __restrict__ clsLDS,
    const float (&fc)[CF], float (&wsacc)[NC + 1],
    int h, int w, int cy, int cx) {
  for (int dy = -HALO; dy <= HALO; ++dy) {
    for (int dx = -HALO; dx <= HALO; ++dx) {
      const int d2 = dy * dy + dx * dx;
      if (d2 >= 25) continue;            // radius-5 Euclidean disc (incl. 0,0)
      if (CHECK) {
        int hn = h + dy, wn = w + dx;
        if (hn < 0 || hn >= HH || wn < 0 || wn >= WW) continue;
      }
      const int nidx = (cy + dy) * HT + (cx + dx);
      // ---- feature distance: 4x 16B vector LDS loads ----
      const h8* fr = (const h8*)(featLDS + nidx * CF);
      float fd2 = 0.0f;
      #pragma unroll
      for (int j = 0; j < 4; ++j) {
        h8 v = fr[j];
        #pragma unroll
        for (int e = 0; e < 8; ++e) {
          float df = fc[j * 8 + e] - (float)v[e];
          fd2 += df * df;
        }
      }
      // sigma_dist=4 -> 1/sd^2 = 1/16 ; sigma_feat=1 -> 1/sf^2 = 1
      float wgt = __expf((float)(-d2) * 0.0625f - fd2);
      // ---- neighbor classification: 3x 16B vector LDS loads ----
      const h8* sn = (const h8*)(clsLDS + nidx * CS);
      h8 s0 = sn[0], s1 = sn[1], s2 = sn[2];
      #pragma unroll
      for (int e = 0; e < 8; ++e) wsacc[e] += wgt * (float)s0[e];
      #pragma unroll
      for (int e = 0; e < 8; ++e) wsacc[8 + e] += wgt * (float)s1[e];
      #pragma unroll
      for (int e = 0; e < 5; ++e) wsacc[16 + e] += wgt * (float)s2[e];
      wsacc[NC] += wgt;                  // degree
    }
  }
}

__global__ __launch_bounds__(256)
void ncut_tile_kernel(const float* __restrict__ cls,
                      const float* __restrict__ feat,
                      float* __restrict__ gnum) {
  __shared__ __align__(32) char smem[SMEM_BYTES];
  _Float16* clsLDS  = (_Float16*)smem;
  _Float16* featLDS = (_Float16*)(smem + FEAT_OFF);

  const int tid = threadIdx.x;            // 0..255
  const int tx  = tid & 15, ty = tid >> 4;
  const int bx = blockIdx.x, by = blockIdx.y, b = blockIdx.z;
  const int h = by * TILE + ty;
  const int w = bx * TILE + tx;

  // ---- Halo load: 24x24 pixels of features + classification -> f16 LDS ----
  for (int i = tid; i < HT * HT; i += 256) {
    int hy = i / HT, wx = i - hy * HT;
    int gh = by * TILE + hy - HALO;
    int gw = bx * TILE + wx - HALO;
    if (gh >= 0 && gh < HH && gw >= 0 && gw < WW) {
      unsigned pix = (unsigned)gh * WW + (unsigned)gw;
      unsigned fb  = (unsigned)b * CF * (HH * WW) + pix;
      #pragma unroll
      for (int c = 0; c < CF; ++c)
        featLDS[i * CF + c] = (_Float16)feat[fb + (unsigned)c * (HH * WW)];
      unsigned cb  = (unsigned)b * NC * (HH * WW) + pix;
      #pragma unroll
      for (int k = 0; k < NC; ++k)
        clsLDS[i * CS + k] = (_Float16)cls[cb + (unsigned)k * (HH * WW)];
    } else {
      #pragma unroll
      for (int c = 0; c < CF; ++c) featLDS[i * CF + c] = (_Float16)0.0f;
      #pragma unroll
      for (int k = 0; k < NC; ++k) clsLDS[i * CS + k] = (_Float16)0.0f;
    }
    #pragma unroll
    for (int k = NC; k < CS; ++k) clsLDS[i * CS + k] = (_Float16)0.0f;  // pads
  }
  __syncthreads();

  // ---- Per-pixel affinity accumulation over 69 offsets ----
  const int cy = ty + HALO, cx = tx + HALO;
  float fc[CF];
  {
    const h8* fr = (const h8*)(featLDS + (cy * HT + cx) * CF);
    #pragma unroll
    for (int j = 0; j < 4; ++j) {
      h8 v = fr[j];
      #pragma unroll
      for (int e = 0; e < 8; ++e) fc[j * 8 + e] = (float)v[e];
    }
  }
  float wsacc[NC + 1];                   // 0..20 = WS, 21 = degree
  #pragma unroll
  for (int k = 0; k <= NC; ++k) wsacc[k] = 0.0f;

  const bool interior = (bx > 0) && (bx < WW / TILE - 1) &&
                        (by > 0) && (by < HH / TILE - 1);
  if (interior)
    affinity_loop<false>(featLDS, clsLDS, fc, wsacc, h, w, cy, cx);
  else
    affinity_loop<true>(featLDS, clsLDS, fc, wsacc, h, w, cy, cx);

  // Snapshot own classification before the halo region is overlaid.
  float sc[NC];
  {
    const h8* sr = (const h8*)(clsLDS + (cy * HT + cx) * CS);
    h8 s0 = sr[0], s1 = sr[1], s2 = sr[2];
    #pragma unroll
    for (int e = 0; e < 8; ++e) sc[e] = (float)s0[e];
    #pragma unroll
    for (int e = 0; e < 8; ++e) sc[8 + e] = (float)s1[e];
    #pragma unroll
    for (int e = 0; e < 5; ++e) sc[16 + e] = (float)s2[e];
  }
  __syncthreads();                       // all halo reads complete

  // ---- Stage WMMA fragments in native per-lane layout ----
  // sAfrag/sBfrag: [wave][tileHalf][lane][16 halves]; each lane's 16 halves
  // are contiguous (32B) so operands load as 2x ds_load_b128, no merges.
  _Float16* sA = (_Float16*)smem;                    // 16 KB
  _Float16* sB = (_Float16*)(smem + 16384);          // 16 KB
  const int lane = tid & 31;
  const int wv   = tid >> 5;
  const int q    = lane;                 // pixel index within the wave's 32
  {
    // A staging: value(class c, pixel q) -> lane (c%16)+16*hiA, elem iA
    const int hiA = (q >> 3) & 1;
    const int iA  = (q & 7) + 8 * (q >> 4);
    _Float16* aw = sA + wv * (2 * 32 * 16);
    #pragma unroll
    for (int c = 0; c < KPAD; ++c) {
      _Float16 v = (c < NC) ? (_Float16)sc[c] : (_Float16)0.0f;
      aw[((c >> 4) * 32 + (c & 15) + 16 * hiA) * 16 + iA] = v;
    }
    // B staging: value(pixel q, class c) -> lane (c%16)+16*hiB, elem iB
    const int hiB = (q >> 3) & 1;
    const int iB  = (q & 7) + 8 * (q >> 4);
    _Float16* bw = sB + wv * (2 * 32 * 16);
    #pragma unroll
    for (int c = 0; c < KPAD; ++c) {
      _Float16 v = (c <= NC) ? (_Float16)wsacc[c] : (_Float16)0.0f;
      bw[((c >> 4) * 32 + (c & 15) + 16 * hiB) * 16 + iB] = v;
    }
  }
  __syncthreads();

  // ---- Per-wave Gram: num[32x32] += S(32cls x 32pix) x [WS|deg](32pix x 32cls)
  const int lm = lane & 15;
  const int hi = lane >> 4;
  const _Float16* aBase = sA + wv * (2 * 32 * 16) + lane * 16;
  const _Float16* bBase = sB + wv * (2 * 32 * 16) + lane * 16;
  v16h A0 = *(const v16h*)(aBase);             // moT = 0 (classes 0..15)
  v16h A1 = *(const v16h*)(aBase + 512);       // moT = 1 (classes 16..31)
  v16h B0 = *(const v16h*)(bBase);             // noT = 0
  v16h B1 = *(const v16h*)(bBase + 512);       // noT = 1

  float* gb = gnum + b * (KPAD * KPAD);
  #pragma unroll
  for (int t = 0; t < 4; ++t) {
    const int mo = (t >> 1) * 16;
    const int no = (t & 1) * 16;
    v16h A = (t >> 1) ? A1 : A0;
    v16h Bf = (t & 1) ? B1 : B0;
    v8f acc = {};
    acc = __builtin_amdgcn_wmma_f32_16x16x32_f16(
        /*neg_a=*/false, A, /*neg_b=*/false, Bf,
        /*c_mod=*/(short)0, acc, /*reuse_a=*/false, /*reuse_b=*/false);
    // D layout: VGPR r -> M = r + 8*hi, N = lane%16.
    #pragma unroll
    for (int r = 0; r < 8; ++r) {
      int M = mo + r + hi * 8;
      int N = no + lm;
      atomicAdd(&gb[M * KPAD + N], acc[r]);
    }
  }
}

__global__ __launch_bounds__(256)
void ncut_finalize_kernel(const float* __restrict__ gnum,
                          float* __restrict__ out) {
  __shared__ float red[256];
  float acc = 0.0f;
  for (int i = threadIdx.x; i < BB * NC * NC; i += 256) {
    int b = i / (NC * NC);
    int r = i - b * (NC * NC);
    int k = r / NC;
    int j = r - k * NC;
    const float* nb = gnum + b * (KPAD * KPAD);
    // num[b,k,j] / den[b,j] ; den lives in column 21 of row j.
    acc += nb[k * KPAD + j] / nb[j * KPAD + NC];
  }
  red[threadIdx.x] = acc;
  __syncthreads();
  for (int s = 128; s > 0; s >>= 1) {
    if (threadIdx.x < s) red[threadIdx.x] += red[threadIdx.x + s];
    __syncthreads();
  }
  if (threadIdx.x == 0) out[0] = red[0];
}

extern "C" void kernel_launch(void* const* d_in, const int* in_sizes, int n_in,
                              void* d_out, int out_size, void* d_ws, size_t ws_size,
                              hipStream_t stream) {
  const float* cls  = (const float*)d_in[0];  // [8,21,256,256]
  const float* feat = (const float*)d_in[1];  // [8,32,256,256]
  float* gnum = (float*)d_ws;                 // [8][32][32] accumulators (32 KB)

  ncut_zero_kernel<<<(BB * KPAD * KPAD + 255) / 256, 256, 0, stream>>>(
      gnum, BB * KPAD * KPAD);

  dim3 grid(WW / TILE, HH / TILE, BB);        // 16 x 16 x 8 tiles
  ncut_tile_kernel<<<grid, 256, 0, stream>>>(cls, feat, gnum);

  ncut_finalize_kernel<<<1, 256, 0, stream>>>(gnum, (float*)d_out);
}
